// graphtransformer_model_paper_mb_32212254720360
// MI455X (gfx1250) — matrix operand order repaired and checked by reading, not compiled, not run on hardware
//
#include <hip/hip_runtime.h>
#include <hip/hip_bf16.h>

#define NU 50000
#define NI 50000
#define NE 100000
#define KDIM 256
#define ODIM 512
#define NH 8
#define HD 64

typedef __attribute__((ext_vector_type(16))) __bf16 v16bf;
typedef __attribute__((ext_vector_type(8)))  float  v8f;

// ---------------------------------------------------------------- utilities
__device__ __forceinline__ unsigned encOrd(float f) {
    unsigned u = __float_as_uint(f);
    return (u & 0x80000000u) ? ~u : (u | 0x80000000u);
}
__device__ __forceinline__ float decOrd(unsigned u) {
    return (u & 0x80000000u) ? __uint_as_float(u ^ 0x80000000u)
                             : __uint_as_float(~u);
}

// ------------------------------------------------------------ f32 -> bf16
__global__ void cvt_bf16(const float* __restrict__ in, __bf16* __restrict__ out, int n) {
    int i = blockIdx.x * blockDim.x + threadIdx.x;
    if (i < n) out[i] = (__bf16)in[i];
}

// ------------------------------------------------------------ fused GEMM
// z = 0..5 : {q_user, k_user, v_user, q_item, k_item, v_item}
// q/k: fused epilogue  attn[n,h] += y[n,o] * wa[o%64]   (never materialized)
// v  : full store to v buffer [N,512] f32
struct GemmArgs {
    const __bf16* xb[2];      // bf16 activations: user, item
    const __bf16* Wb[6];      // bf16 weights, [512,256] row-major each
    const float*  bias[6];
    float*        vout[2];    // v_user, v_item
    float*        attn[6];    // aq_u, ak_u, 0, aq_i, ak_i, 0
    const float*  wa[2];      // wa_user, wa_item
};

__global__ __launch_bounds__(128) void gemm_qkv(GemmArgs ga) {
    const int z     = blockIdx.z;
    const int t     = z / 3;            // node type
    const int mkind = z % 3;            // 0=q 1=k 2=v
    const int lane  = threadIdx.x & 31;
    const int wave  = threadIdx.x >> 5;
    const int rowBase = blockIdx.x * 16;
    const int nBase   = (blockIdx.y * 4 + wave) * 16;
    const int half = lane >> 4;         // 0 or 1
    const int lid  = lane & 15;
    const int nGlob = nBase + lid;
    const int aRow  = rowBase + lid;

    // A fragment: lane holds row m, 16 contiguous K (lanes 16-31: +16), same order as B
    const __bf16* aPtr = ga.xb[t] + (size_t)aRow * KDIM + half * 16;
    // B fragment: lane holds col n, 16 contiguous K, same order as A
    const __bf16* bPtr = ga.Wb[z] + (size_t)nGlob * KDIM + half * 16;

    __builtin_prefetch(aPtr + 128, 0, 1);   // global_prefetch_b8
    __builtin_prefetch(bPtr + 128, 0, 1);

    const float bsc = ga.bias[z][nGlob];    // bias is column-constant
    v8f c;
#pragma unroll
    for (int i = 0; i < 8; ++i) c[i] = bsc;

#pragma unroll
    for (int kk = 0; kk < KDIM / 32; ++kk) {
        union { v16bf v; uint4 q[2]; } A, B;
        A.q[0] = *(const uint4*)(aPtr + kk * 32);
        A.q[1] = *(const uint4*)(aPtr + kk * 32 + 8);
        B.q[0] = *(const uint4*)(bPtr + kk * 32);
        B.q[1] = *(const uint4*)(bPtr + kk * 32 + 8);
        c = __builtin_amdgcn_wmma_f32_16x16x32_bf16(
                false, A.v, false, B.v, (short)0, c, false, false);
    }

    const int mOff = half * 8;            // C layout: VGPR i -> M = i + 8*half
    if (mkind == 2) {
        float* out = ga.vout[t] + nGlob;
#pragma unroll
        for (int i = 0; i < 8; ++i)
            out[(size_t)(rowBase + mOff + i) * ODIM] = c[i];
    } else {
        const int head = nGlob >> 6, d = nGlob & 63;
        const float w = ga.wa[t][d];
        float* attn = ga.attn[z];
#pragma unroll
        for (int i = 0; i < 8; ++i)
            atomicAdd(attn + (size_t)(rowBase + mOff + i) * NH + head, c[i] * w);
    }
}

// ---------------------------------------------------- edge pass 1: score + max + deg
__global__ void edge_score_max(const int* __restrict__ src, const int* __restrict__ dst,
                               const float* __restrict__ akS, const float* __restrict__ aqD,
                               float* __restrict__ score, unsigned* __restrict__ mEnc,
                               float* __restrict__ deg) {
    int idx = blockIdx.x * blockDim.x + threadIdx.x;
    if (idx >= NE * NH) return;
    int e = idx >> 3, h = idx & 7;
    int s = src[e], d = dst[e];
    float sc = (akS[s * NH + h] + aqD[d * NH + h]) * 0.125f;   // 1/sqrt(64)
    score[idx] = sc;
    atomicMax(mEnc + d * NH + h, encOrd(sc));
    if (h == 0) atomicAdd(deg + d, 1.0f);
}

// ---------------------------------------------------- edge pass 2: exp + denom
__global__ void edge_expsum(const int* __restrict__ dst, float* __restrict__ score,
                            const unsigned* __restrict__ mEnc, float* __restrict__ den) {
    int idx = blockIdx.x * blockDim.x + threadIdx.x;
    if (idx >= NE * NH) return;
    int e = idx >> 3, h = idx & 7;
    int d = dst[e];
    float ex = __expf(score[idx] - decOrd(mEnc[d * NH + h]));
    score[idx] = ex;
    atomicAdd(den + d * NH + h, ex);
}

// ---------------------------------------------------- edge pass 3: weighted scatter-add
// fuses attn normalization, /max(deg,1) mean, and inter-type sum (accumulates into d_out)
__global__ void edge_message(const int* __restrict__ src, const int* __restrict__ dst,
                             const float* __restrict__ ex, const float* __restrict__ den,
                             const float* __restrict__ deg, const float* __restrict__ v,
                             float* __restrict__ out) {
    int idx = blockIdx.x * blockDim.x + threadIdx.x;
    if (idx >= NE * ODIM) return;
    int e = idx >> 9, r = idx & 511, h = r >> 6;
    int s = src[e], d = dst[e];
    float attn = ex[e * NH + h] / den[d * NH + h];
    float rd = 1.0f / fmaxf(deg[d], 1.0f);
    atomicAdd(out + (size_t)d * ODIM + r, v[(size_t)s * ODIM + r] * attn * rd);
}

// ================================================================= launch
extern "C" void kernel_launch(void* const* d_in, const int* in_sizes, int n_in,
                              void* d_out, int out_size, void* d_ws, size_t ws_size,
                              hipStream_t stream) {
    const float* x_user  = (const float*)d_in[0];
    const float* x_item  = (const float*)d_in[1];
    const float* W6[6]   = { (const float*)d_in[2], (const float*)d_in[4],
                             (const float*)d_in[6], (const float*)d_in[8],
                             (const float*)d_in[10], (const float*)d_in[12] };
    const float* B6[6]   = { (const float*)d_in[3], (const float*)d_in[5],
                             (const float*)d_in[7], (const float*)d_in[9],
                             (const float*)d_in[11], (const float*)d_in[13] };
    const float* wa_user = (const float*)d_in[14];
    const float* wa_item = (const float*)d_in[15];
    const int* srcE[3] = { (const int*)d_in[16], (const int*)d_in[18], (const int*)d_in[20] };
    const int* dstE[3] = { (const int*)d_in[17], (const int*)d_in[19], (const int*)d_in[21] };

    // ---- workspace carve-up (256B aligned)
    char* ws = (char*)d_ws;
    size_t off = 0;
    auto carve = [&](size_t bytes) -> char* {
        char* p = ws + off;
        off = (off + bytes + 255) & ~(size_t)255;
        return p;
    };
    __bf16* xbU = (__bf16*)carve((size_t)NU * KDIM * 2);
    __bf16* xbI = (__bf16*)carve((size_t)NI * KDIM * 2);
    __bf16* Wb  = (__bf16*)carve((size_t)6 * ODIM * KDIM * 2);
    float*  vU  = (float*)carve((size_t)NU * ODIM * 4);
    float*  vI  = (float*)carve((size_t)NI * ODIM * 4);

    size_t zStart = off;                      // everything below is zero-initialized
    float* aqU = (float*)carve((size_t)NU * NH * 4);
    float* akU = (float*)carve((size_t)NU * NH * 4);
    float* aqI = (float*)carve((size_t)NI * NH * 4);
    float* akI = (float*)carve((size_t)NI * NH * 4);
    unsigned* mE[3];  float* denE[3];  float* degE[3];
    const int NdE[3] = { NI, NU, NU };
    for (int i = 0; i < 3; ++i) {
        mE[i]   = (unsigned*)carve((size_t)NdE[i] * NH * 4);
        denE[i] = (float*)carve((size_t)NdE[i] * NH * 4);
        degE[i] = (float*)carve((size_t)NdE[i] * 4);
    }
    size_t zEnd = off;
    float* exE[3];
    for (int i = 0; i < 3; ++i) exE[i] = (float*)carve((size_t)NE * NH * 4);
    (void)ws_size; (void)in_sizes; (void)n_in;

    hipMemsetAsync(ws + zStart, 0, zEnd - zStart, stream);
    hipMemsetAsync(d_out, 0, (size_t)out_size * sizeof(float), stream);

    // ---- bf16 conversion
    {
        int n = NU * KDIM;
        cvt_bf16<<<(n + 255) / 256, 256, 0, stream>>>(x_user, xbU, n);
        n = NI * KDIM;
        cvt_bf16<<<(n + 255) / 256, 256, 0, stream>>>(x_item, xbI, n);
        n = ODIM * KDIM;
        for (int i = 0; i < 6; ++i)
            cvt_bf16<<<(n + 255) / 256, 256, 0, stream>>>(W6[i], Wb + (size_t)i * n, n);
    }

    // ---- WMMA GEMMs (q/k fused into attn scalars; v stored)
    GemmArgs ga;
    ga.xb[0] = xbU;  ga.xb[1] = xbI;
    for (int i = 0; i < 6; ++i) { ga.Wb[i] = Wb + (size_t)i * ODIM * KDIM; ga.bias[i] = B6[i]; }
    ga.vout[0] = vU; ga.vout[1] = vI;
    ga.attn[0] = aqU; ga.attn[1] = akU; ga.attn[2] = nullptr;
    ga.attn[3] = aqI; ga.attn[4] = akI; ga.attn[5] = nullptr;
    ga.wa[0] = wa_user; ga.wa[1] = wa_item;
    gemm_qkv<<<dim3(NU / 16, ODIM / 16 / 4, 6), 128, 0, stream>>>(ga);

    // ---- edge passes
    float* h_user = (float*)d_out;
    float* h_item = (float*)d_out + (size_t)NU * ODIM;
    const float* akS[3] = { akU, akI, akU };     // clicks, clicked_by, follows
    const float* aqD[3] = { aqI, aqU, aqU };
    const float* vS[3]  = { vU,  vI,  vU  };
    float*       oD[3]  = { h_item, h_user, h_user };

    const int gEH = (NE * NH + 255) / 256;
    const int gEO = (NE * ODIM + 255) / 256;
    for (int i = 0; i < 3; ++i) {
        edge_score_max<<<gEH, 256, 0, stream>>>(srcE[i], dstE[i], akS[i], aqD[i],
                                                exE[i], mE[i], degE[i]);
        edge_expsum<<<gEH, 256, 0, stream>>>(dstE[i], exE[i], mE[i], denE[i]);
        edge_message<<<gEO, 256, 0, stream>>>(srcE[i], dstE[i], exE[i], denE[i],
                                              degE[i], vS[i], oD[i]);
    }
}
